// Decoder_70437463654609
// MI455X (gfx1250) — matrix operand order repaired and checked by reading, not compiled, not run on hardware
//
#include <hip/hip_runtime.h>
#include <hip/hip_bf16.h>
#include <math.h>

// ---------------------------------------------------------------------------
// MI455X (gfx1250, wave32) LSTM decoder.
//
//  - Persistent tiling: block b owns batch rows [16b, 16b+16); the whole
//    32-step recurrence runs inside one workgroup (rows are independent).
//  - Heavy matmul h@w_hh^T via v_wmma_f32_16x16x32_f16 (f16 in / f32 acc).
//  - x-path folded:  x@w_ih^T == relu_bn(lp@se_w1..) @ (se_w2@w_ih^T) + const,
//    so the 512-wide x matmul becomes a single K=16(padded 32) WMMA.
//  - 512 threads = 16 waves; each wave owns 32 j-columns and all 4 gates for
//    them -> LSTM update is wave-local; c stays in VGPRs for all 32 steps.
//  - asm memory clobbers keep the (loop-invariant) weight-tile loads inside
//    the step loop; without them LICM hoists ~4KB/lane of tiles -> scratch
//    spill catastrophe (observed in round 1).
// ---------------------------------------------------------------------------

typedef __attribute__((ext_vector_type(16))) _Float16 v16h;
typedef __attribute__((ext_vector_type(8)))  float    v8f;

union Frag32 { uint4 u[2]; v16h h; };        // 32B: one WMMA A/B per-lane fragment
union H8     { uint4 u;    _Float16 e[8]; }; // 16B: 8 halves

#define B_TOT   2048
#define H_DIM   512
#define NG      2048          // 4*H
#define SEQ     32
#define HSTR    528           // padded f16 row stride for h in LDS (16B-aligned rows)
#define NTHREADS 512          // 16 waves

// workspace layout (bytes)
#define WHH_OFF   ((size_t)0)                    // f16, 2048*512   = 2 MiB (tile-packed)
#define WP_OFF    ((size_t)2097152)              // f16, 128t*32*16 = 128 KiB
#define BIAS_OFF  ((size_t)2228224)              // f32, 2048
#define SEC_OFF   ((size_t)2236416)              // f32, 48  (A1,A2,A0)
#define HPC_OFF   ((size_t)2236608)              // f32, 32  (scale, off)
#define W1T_OFF   ((size_t)2236736)              // f32, 16*512 (hp_w1 transposed)

__device__ __forceinline__ float sigm(float x) { return 1.f / (1.f + __expf(-x)); }

__device__ __forceinline__ Frag32 ldg_frag(const _Float16* p) {
    Frag32 f;
    const uint4* q = (const uint4*)p;
    f.u[0] = q[0];
    f.u[1] = q[1];
    return f;
}

// ---------------- pack kernels (cheap, run every launch) -------------------

// w_hh [2048][512] f32 -> WMMA-B tile-packed f16.
// tile = nt*16 + kt ; per tile 32 lanes x 16 f16.
// B layout (16-bit, 32x16): N=lane%16; K = 8*(lane/16) + slot (slot<8), 16 + 8*(lane/16) + slot-8 (slot>=8).
__global__ void pack_whh(const float* __restrict__ w_hh, _Float16* __restrict__ out) {
    int idx = blockIdx.x * 256 + threadIdx.x;           // 0 .. 2048*512-1
    int slot = idx & 15;
    int lane = (idx >> 4) & 31;
    int tile = idx >> 9;
    int kt = tile & 15;
    int nt = tile >> 4;
    int n = nt * 16 + (lane & 15);
    int k = kt * 32 + ((slot >> 3) << 4) + ((lane >> 4) << 3) + (slot & 7);
    out[idx] = (_Float16)w_hh[(size_t)n * H_DIM + k];
}

// Wp = se_w2 @ w_ih^T  ([16 x 2048]), packed as one K-tile (K padded 16->32).
__global__ void pack_wp(const float* __restrict__ se_w2, const float* __restrict__ w_ih,
                        _Float16* __restrict__ out) {
    int idx = blockIdx.x * 256 + threadIdx.x;           // 0 .. 128*32*16-1
    int slot = idx & 15;
    int lane = (idx >> 4) & 31;
    int nt = idx >> 9;
    int n = nt * 16 + (lane & 15);
    float v = 0.f;
    if (slot < 8) {                                     // K = 8*(lane/16) + slot in [0,16): valid
        const float* a = se_w2 + (size_t)(((lane >> 4) << 3) + slot) * 512;
        const float* b = w_ih + (size_t)n * 512;
        for (int e = 0; e < 512; ++e) v = fmaf(a[e], b[e], v);
    }                                                   // slot>=8 -> K in [16,32): zero pad
    out[idx] = (_Float16)v;
}

// biasc[n] = b_ih[n] + b_hh[n] + se_b2 @ w_ih[n,:]
__global__ void pack_bias(const float* __restrict__ se_b2, const float* __restrict__ w_ih,
                          const float* __restrict__ b_ih, const float* __restrict__ b_hh,
                          float* __restrict__ out) {
    int n = blockIdx.x * 256 + threadIdx.x;
    if (n >= NG) return;
    float v = b_ih[n] + b_hh[n];
    const float* b = w_ih + (size_t)n * 512;
    for (int e = 0; e < 512; ++e) v = fmaf(se_b2[e], b[e], v);
    out[n] = v;
}

// Fold BN affine into per-feature scale/offset for both small MLPs.
__global__ void pack_small(const float* se_w1, const float* se_b1, const float* se_g,
                           const float* se_bt, const float* se_m, const float* se_v,
                           const float* hp_b1, const float* hp_g, const float* hp_bt,
                           const float* hp_m, const float* hp_v,
                           float* __restrict__ seC, float* __restrict__ hpC) {
    int t = threadIdx.x;
    if (t < 16) {
        float s = se_g[t] * rsqrtf(se_v[t] + 1e-5f);
        seC[t]      = se_w1[t] * s;                       // A1 (w1[0][j])
        seC[16 + t] = se_w1[16 + t] * s;                  // A2 (w1[1][j])
        seC[32 + t] = (se_b1[t] - se_m[t]) * s + se_bt[t];// A0
    } else if (t < 32) {
        int j = t - 16;
        float s = hp_g[j] * rsqrtf(hp_v[j] + 1e-5f);
        hpC[j]      = s;
        hpC[16 + j] = (hp_b1[j] - hp_m[j]) * s + hp_bt[j];
    }
}

// hp_w1 [512][16] -> [16][512] for broadcast-friendly reads.
__global__ void pack_w1t(const float* __restrict__ hp_w1, float* __restrict__ out) {
    int idx = blockIdx.x * 256 + threadIdx.x;           // 0..8191
    int j = idx >> 9, k = idx & 511;
    out[idx] = hp_w1[(size_t)k * 16 + j];
}

// ---------------------------- main kernel ----------------------------------

__global__ void __launch_bounds__(NTHREADS)
lstm_decoder(const float* __restrict__ last_pos, const float* __restrict__ hh,
             const float* __restrict__ ch, const float* __restrict__ hp_w2,
             const float* __restrict__ hp_b2,
             const _Float16* __restrict__ whhP, const _Float16* __restrict__ wpP,
             const float* __restrict__ biasc, const float* __restrict__ seC,
             const float* __restrict__ hpC, const float* __restrict__ w1t,
             float* __restrict__ out) {
    __shared__ _Float16 hF[16 * HSTR];   // h tile, f16 row-major (padded stride)
    __shared__ _Float16 aX[32 * 16];     // relu_x A-fragment (K padded 16->32)
    __shared__ float hpbuf[16 * 16];
    __shared__ float lpS[32];
    __shared__ float sA1[16], sA2[16], sA0[16], hSc[16], hOf[16];

    const int t = threadIdx.x;
    const int lane = t & 31;
    const int wave = t >> 5;             // 16 waves; wave owns j in [wave*32, wave*32+32)
    const int lrow = lane & 15;
    const int lhalf = lane >> 4;
    const int wgbase = blockIdx.x * 16;

    if (t < 16) {
        sA1[t] = seC[t]; sA2[t] = seC[16 + t]; sA0[t] = seC[32 + t];
        hSc[t] = hpC[t]; hOf[t] = hpC[16 + t];
        lpS[t * 2]     = last_pos[(size_t)(wgbase + t) * 2];
        lpS[t * 2 + 1] = last_pos[(size_t)(wgbase + t) * 2 + 1];
    }
    if (t < 32) {                         // K-padding of aX never changes: zero once
        #pragma unroll
        for (int s = 8; s < 16; ++s) aX[t * 16 + s] = (_Float16)0.f;
    }
    for (int idx = t; idx < 16 * 512; idx += NTHREADS) { // h0 -> f16 LDS
        int m = idx >> 9, k = idx & 511;
        hF[m * HSTR + k] = (_Float16)hh[(size_t)(wgbase + m) * H_DIM + k];
    }

    // c0 into registers with the WMMA C layout (lane: n = lane%16, m = v + 8*(lane>=16))
    v8f creg[2];
    #pragma unroll
    for (int jt = 0; jt < 2; ++jt) {
        int j = wave * 32 + jt * 16 + lrow;
        #pragma unroll
        for (int v = 0; v < 8; ++v)
            creg[jt][v] = ch[(size_t)(wgbase + v + lhalf * 8) * H_DIM + j];
    }

    // per-lane gate biases (constant across steps)
    float biasS[8];
    #pragma unroll
    for (int g = 0; g < 4; ++g)
        #pragma unroll
        for (int jt = 0; jt < 2; ++jt)
            biasS[g * 2 + jt] = biasc[g * 512 + wave * 32 + jt * 16 + lrow];

    v8f acc[8];

    for (int step = 0; step < SEQ; ++step) {
        __syncthreads();                                  // hF / lpS ready

        // ---- SE MLP (folded): relu_x -> packed A fragment, K padded to 32
        if (t < 16) {
            float l0 = lpS[t * 2], l1 = lpS[t * 2 + 1];
            #pragma unroll
            for (int j = 0; j < 16; ++j) {
                float r = fmaxf(0.f, fmaf(l0, sA1[j], fmaf(l1, sA2[j], sA0[j])));
                // A layout: lanes 0-15 hold K=0-7 in slots 0-7; lanes 16-31 hold K=8-15
                if (j < 8) aX[t * 16 + j]              = (_Float16)r;
                else       aX[(t + 16) * 16 + (j - 8)] = (_Float16)r;
            }
        }
        __syncthreads();

        // keep weight loads from being hoisted out of the step loop (LICM ->
        // scratch-spill disaster otherwise: weights are const/invariant)
        asm volatile("" ::: "memory");

        // ---- init acc = bias, then the folded x-path as one WMMA per n-tile
        Frag32 ax;
        { const uint4* p = (const uint4*)(aX + lane * 16); ax.u[0] = p[0]; ax.u[1] = p[1]; }
        #pragma unroll
        for (int i = 0; i < 8; ++i)
            #pragma unroll
            for (int v = 0; v < 8; ++v) acc[i][v] = biasS[i];

        {
            Frag32 b[8];
            #pragma unroll
            for (int g = 0; g < 4; ++g)
                #pragma unroll
                for (int jt = 0; jt < 2; ++jt) {
                    int i = g * 2 + jt;
                    int nt = g * 32 + wave * 2 + jt;
                    b[i] = ldg_frag(wpP + ((size_t)nt * 32 + lane) * 16);
                }
            #pragma unroll
            for (int i = 0; i < 8; ++i)
                acc[i] = __builtin_amdgcn_wmma_f32_16x16x32_f16(
                    false, ax.h, false, b[i].h, (short)0, acc[i], false, false);
        }

        // ---- main recurrence matmul: gates += h @ w_hh^T  (K = 512)
        for (int kt = 0; kt < 16; ++kt) {
            asm volatile("" ::: "memory");                // see note above
            Frag32 a;
            const _Float16* ar = hF + lrow * HSTR + kt * 32 + lhalf * 8;
            a.u[0] = *(const uint4*)ar;          // K 0-7  (or 8-15)
            a.u[1] = *(const uint4*)(ar + 16);   // K 16-23(or 24-31)
            Frag32 b[8];
            #pragma unroll
            for (int g = 0; g < 4; ++g)
                #pragma unroll
                for (int jt = 0; jt < 2; ++jt) {
                    int i = g * 2 + jt;
                    int nt = g * 32 + wave * 2 + jt;
                    b[i] = ldg_frag(whhP + (((size_t)nt * 16 + kt) * 32 + lane) * 16);
                }
            #pragma unroll
            for (int i = 0; i < 8; ++i)
                acc[i] = __builtin_amdgcn_wmma_f32_16x16x32_f16(
                    false, a.h, false, b[i].h, (short)0, acc[i], false, false);
        }
        __syncthreads();                                  // all reads of old h done

        // ---- LSTM pointwise (all four gates are wave-local); write new h (f16)
        #pragma unroll
        for (int jt = 0; jt < 2; ++jt) {
            #pragma unroll
            for (int v = 0; v < 8; ++v) {
                float i_ = sigm(acc[0 * 2 + jt][v]);
                float f_ = sigm(acc[1 * 2 + jt][v]);
                float g_ = tanhf(acc[2 * 2 + jt][v]);
                float o_ = sigm(acc[3 * 2 + jt][v]);
                float c = fmaf(f_, creg[jt][v], i_ * g_);
                creg[jt][v] = c;
                float hv = o_ * tanhf(c);
                int mm = v + lhalf * 8;
                int j = wave * 32 + jt * 16 + lrow;
                hF[mm * HSTR + j] = (_Float16)hv;
            }
        }
        __syncthreads();                                  // new h visible

        // ---- head MLP stage 1: [16 x 512] @ [512 x 16], BN-fold + ReLU
        if (t < 256) {
            int m = t & 15, jc = t >> 4;
            const float* wr = w1t + jc * 512;
            float s = 0.f;
            for (int k0 = 0; k0 < 512; k0 += 8) {
                H8 hv; hv.u = *(const uint4*)(hF + m * HSTR + k0);
                #pragma unroll
                for (int q = 0; q < 8; ++q) s = fmaf((float)hv.e[q], wr[k0 + q], s);
            }
            hpbuf[m * 16 + jc] = fmaxf(fmaf(s, hSc[jc], hOf[jc]), 0.f);
        }
        __syncthreads();

        // ---- head MLP stage 2 + sigmoid position update + trajectory store
        if (t < 32) {
            int m = t >> 1, d = t & 1;
            float pos = hp_b2[d];
            #pragma unroll
            for (int j = 0; j < 16; ++j) pos = fmaf(hpbuf[m * 16 + j], hp_w2[j * 2 + d], pos);
            float lpn = sigm(pos + lpS[m * 2 + d]);
            lpS[m * 2 + d] = lpn;
            out[((size_t)step * B_TOT + (wgbase + m)) * 2 + d] = lpn;
        }
    }
}

// ---------------------------------------------------------------------------

extern "C" void kernel_launch(void* const* d_in, const int* in_sizes, int n_in,
                              void* d_out, int out_size, void* d_ws, size_t ws_size,
                              hipStream_t stream) {
    const float* last_pos = (const float*)d_in[0];
    const float* hh    = (const float*)d_in[1];
    const float* ch    = (const float*)d_in[2];
    const float* se_w1 = (const float*)d_in[3];
    const float* se_b1 = (const float*)d_in[4];
    const float* se_g  = (const float*)d_in[5];
    const float* se_bt = (const float*)d_in[6];
    const float* se_m  = (const float*)d_in[7];
    const float* se_v  = (const float*)d_in[8];
    const float* se_w2 = (const float*)d_in[9];
    const float* se_b2 = (const float*)d_in[10];
    const float* w_ih  = (const float*)d_in[11];
    const float* w_hh  = (const float*)d_in[12];
    const float* b_ih  = (const float*)d_in[13];
    const float* b_hh  = (const float*)d_in[14];
    const float* hp_w1 = (const float*)d_in[15];
    const float* hp_b1 = (const float*)d_in[16];
    const float* hp_g  = (const float*)d_in[17];
    const float* hp_bt = (const float*)d_in[18];
    const float* hp_m  = (const float*)d_in[19];
    const float* hp_v  = (const float*)d_in[20];
    const float* hp_w2 = (const float*)d_in[21];
    const float* hp_b2 = (const float*)d_in[22];

    char* ws = (char*)d_ws;
    _Float16* whhP = (_Float16*)(ws + WHH_OFF);
    _Float16* wpP  = (_Float16*)(ws + WP_OFF);
    float* biasC   = (float*)(ws + BIAS_OFF);
    float* seC     = (float*)(ws + SEC_OFF);
    float* hpC     = (float*)(ws + HPC_OFF);
    float* w1t     = (float*)(ws + W1T_OFF);

    pack_whh<<<(NG * H_DIM) / 256, 256, 0, stream>>>(w_hh, whhP);
    pack_wp<<<(128 * 32 * 16) / 256, 256, 0, stream>>>(se_w2, w_ih, wpP);
    pack_bias<<<NG / 256, 256, 0, stream>>>(se_b2, w_ih, b_ih, b_hh, biasC);
    pack_small<<<1, 64, 0, stream>>>(se_w1, se_b1, se_g, se_bt, se_m, se_v,
                                     hp_b1, hp_g, hp_bt, hp_m, hp_v, seC, hpC);
    pack_w1t<<<(16 * 512) / 256, 256, 0, stream>>>(hp_w1, w1t);

    lstm_decoder<<<B_TOT / 16, NTHREADS, 0, stream>>>(
        last_pos, hh, ch, hp_w2, hp_b2, whhP, wpP, biasC, seC, hpC, w1t,
        (float*)d_out);
}
